// Net_43344809951841
// MI455X (gfx1250) — matrix operand order repaired and checked by reading, not compiled, not run on hardware
//
#include <hip/hip_runtime.h>

#define NNODES 50000
#define NEDGES 800000
#define ETOT   (NEDGES + NNODES)
#define NGRAPH 128
#define INF_   768
#define HID2   128
#define NHID   64

typedef __attribute__((ext_vector_type(16))) __bf16       v16bf;
typedef __attribute__((ext_vector_type(8)))  float        v8f;
typedef __attribute__((ext_vector_type(8)))  unsigned int v8u;

__device__ __forceinline__ __bf16 f2bf(float f) {
  unsigned u = __builtin_bit_cast(unsigned, f);
  unsigned r = u + 0x7FFFu + ((u >> 16) & 1u);         // round-to-nearest-even
  unsigned short h = (unsigned short)(r >> 16);
  return __builtin_bit_cast(__bf16, h);
}

__device__ __forceinline__ float selu_f(float x) {
  const float sc = 1.0507009873554805f;
  const float al = 1.6732632423543772f;
  return x > 0.f ? sc * x : sc * al * (expf(x) - 1.f);
}

// monotone float <-> uint mapping for atomicMax on floats
__device__ __forceinline__ unsigned ford(float f) {
  unsigned u = __builtin_bit_cast(unsigned, f);
  return (u & 0x80000000u) ? ~u : (u | 0x80000000u);
}
__device__ __forceinline__ float funord(unsigned v) {
  unsigned u = (v & 0x80000000u) ? (v & 0x7FFFFFFFu) : ~v;
  return __builtin_bit_cast(float, u);
}

__global__ void fill_f32(float* __restrict__ p, float v, int n) {
  int i = blockIdx.x * blockDim.x + threadIdx.x;
  if (i < n) p[i] = v;
}

// Pack W [K x 128] f32 into WMMA-B bf16 layout.
// chunk = (kt*8 + nt)*32 + lane; lane holds col nt*16+(lane&15),
// K = kt*32 + 16*(lane>>4) + e, e = 0..15 (dword j = K{2j,2j+1}).
__global__ void pack_w_bf16(const float* __restrict__ W, __bf16* __restrict__ Wp, int K) {
  int tid = blockIdx.x * blockDim.x + threadIdx.x;
  if (tid >= (K >> 5) * 256) return;
  int kt = tid >> 8, rem = tid & 255;
  int nt = rem >> 5, lane = rem & 31;
  int n = nt * 16 + (lane & 15);
  int kb = kt * 32 + (lane >> 4) * 16;
  __bf16* dst = Wp + ((size_t)(kt * 8 + nt) * 32 + lane) * 16;
#pragma unroll
  for (int e = 0; e < 16; ++e) dst[e] = f2bf(W[(size_t)(kb + e) * HID2 + n]);
}

// Out[M x 128] = X[M x K] @ W (packed bf16), fp32 accumulate via WMMA.
// One wave per 16-row tile; 8 accumulators cover all 128 output columns.
__global__ void gemm_wmma_bf16(const float* __restrict__ X,
                               const __bf16* __restrict__ Wp,
                               float* __restrict__ Out, int M, int K) {
  const int lane = threadIdx.x & 31;
  const int wave = threadIdx.x >> 5;
  const int tile = blockIdx.x * (blockDim.x >> 5) + wave;
  if (tile >= (M >> 4)) return;               // wave-uniform: EXEC stays all-ones
  const int row0 = tile << 4;
  const int half = lane >> 4;
  const int r    = row0 + (lane & 15);
  v8f acc[8] = {};
  const int ktiles = K >> 5;
  for (int kt = 0; kt < ktiles; ++kt) {
    const float* xp = X + (size_t)r * K + kt * 32 + half * 16;
    float4 a0 = *(const float4*)(xp + 0);
    float4 a1 = *(const float4*)(xp + 4);
    float4 a2 = *(const float4*)(xp + 8);
    float4 a3 = *(const float4*)(xp + 12);
    v16bf A;
    A[0]=f2bf(a0.x);  A[1]=f2bf(a0.y);  A[2]=f2bf(a0.z);  A[3]=f2bf(a0.w);
    A[4]=f2bf(a1.x);  A[5]=f2bf(a1.y);  A[6]=f2bf(a1.z);  A[7]=f2bf(a1.w);
    A[8]=f2bf(a2.x);  A[9]=f2bf(a2.y);  A[10]=f2bf(a2.z); A[11]=f2bf(a2.w);
    A[12]=f2bf(a3.x); A[13]=f2bf(a3.y); A[14]=f2bf(a3.z); A[15]=f2bf(a3.w);
#pragma unroll
    for (int nt = 0; nt < 8; ++nt) {
      v8u braw = ((const v8u*)Wp)[(size_t)(kt * 8 + nt) * 32 + lane];
      v16bf B = __builtin_bit_cast(v16bf, braw);
      acc[nt] = __builtin_amdgcn_wmma_f32_16x16x32_bf16(
          false, A, false, B, (short)0, acc[nt], false, false);
    }
  }
#pragma unroll
  for (int nt = 0; nt < 8; ++nt) {
    const int col = nt * 16 + (lane & 15);
#pragma unroll
    for (int rr = 0; rr < 8; ++rr)
      Out[(size_t)(row0 + rr + half * 8) * HID2 + col] = acc[nt][rr];
  }
}

// e_src[n] = xw[n]·a_src, e_dst[n] = xw[n]·a_dst   (one wave per node)
__global__ void node_attn(const float* __restrict__ xw, const float* __restrict__ as,
                          const float* __restrict__ ad, float* __restrict__ es,
                          float* __restrict__ ed, int N) {
  int lane = threadIdx.x & 31, wave = threadIdx.x >> 5;
  int n = blockIdx.x * (blockDim.x >> 5) + wave;
  if (n >= N) return;
  float4 v = ((const float4*)(xw + (size_t)n * HID2))[lane];
  float4 a = ((const float4*)as)[lane];
  float4 b = ((const float4*)ad)[lane];
  float s = v.x * a.x + v.y * a.y + v.z * a.z + v.w * a.w;
  float t = v.x * b.x + v.y * b.y + v.z * b.z + v.w * b.w;
#pragma unroll
  for (int off = 16; off; off >>= 1) {
    s += __shfl_xor(s, off, 32);
    t += __shfl_xor(t, off, 32);
  }
  if (lane == 0) { es[n] = s; ed[n] = t; }
}

__global__ void edge_logits(const float* __restrict__ es, const float* __restrict__ ed,
                            const int* __restrict__ src, const int* __restrict__ dst,
                            float* __restrict__ ee, unsigned* __restrict__ m) {
  int e = blockIdx.x * blockDim.x + threadIdx.x;
  if (e >= ETOT) return;
  int s, d;
  if (e < NEDGES) { s = src[e]; d = dst[e]; } else { s = e - NEDGES; d = s; }
  float v = es[s] + ed[d];
  v = v > 0.f ? v : 0.2f * v;                 // leaky_relu(0.2)
  ee[e] = v;
  atomicMax(m + d, ford(v));
}

__global__ void edge_expsum(const int* __restrict__ src, const int* __restrict__ dst,
                            float* __restrict__ ee, const unsigned* __restrict__ m,
                            float* __restrict__ den) {
  int e = blockIdx.x * blockDim.x + threadIdx.x;
  if (e >= ETOT) return;
  int d = (e < NEDGES) ? dst[e] : (e - NEDGES);
  float num = expf(ee[e] - funord(m[d]));
  ee[e] = num;
  atomicAdd(den + d, num);
}

// out[dst] += (num/den[dst]) * xw[src]  (one wave per edge, 4 feats/lane)
__global__ void edge_scatter(const int* __restrict__ src, const int* __restrict__ dst,
                             const float* __restrict__ ee, const float* __restrict__ den,
                             const float* __restrict__ P, float* __restrict__ Q) {
  int lane = threadIdx.x & 31, wave = threadIdx.x >> 5;
  int e = blockIdx.x * (blockDim.x >> 5) + wave;
  if (e >= ETOT) return;
  int s, d;
  if (e < NEDGES) { s = src[e]; d = dst[e]; } else { s = e - NEDGES; d = s; }
  float alpha = ee[e] / den[d];
  float4 v = ((const float4*)(P + (size_t)s * HID2))[lane];
  float* q = Q + (size_t)d * HID2 + lane * 4;
  atomicAdd(q + 0, alpha * v.x);
  atomicAdd(q + 1, alpha * v.y);
  atomicAdd(q + 2, alpha * v.z);
  atomicAdd(q + 3, alpha * v.w);
}

__global__ void node_bias_selu(float* __restrict__ Q, const float* __restrict__ b, int n) {
  int i = blockIdx.x * blockDim.x + threadIdx.x;
  if (i < n) Q[i] = selu_f(Q[i] + b[i & (HID2 - 1)]);
}

__global__ void pool_sum(const float* __restrict__ h, const int* __restrict__ batch,
                         float* __restrict__ sums, float* __restrict__ cnt, int N) {
  int lane = threadIdx.x & 31, wave = threadIdx.x >> 5;
  int n = blockIdx.x * (blockDim.x >> 5) + wave;
  if (n >= N) return;
  int g = batch[n];
  float4 v = ((const float4*)(h + (size_t)n * HID2))[lane];
  float* o = sums + (size_t)g * HID2 + lane * 4;
  atomicAdd(o + 0, v.x);
  atomicAdd(o + 1, v.y);
  atomicAdd(o + 2, v.z);
  atomicAdd(o + 3, v.w);
  if (lane == 0) atomicAdd(cnt + g, 1.0f);
}

__global__ void pool_final(float* __restrict__ sums, const float* __restrict__ cnt) {
  int i = blockIdx.x * blockDim.x + threadIdx.x;
  if (i < NGRAPH * HID2) sums[i] = selu_f(sums[i] / fmaxf(cnt[i >> 7], 1.0f));
}

__global__ void fc1_kernel(const float* __restrict__ pooled, const float* __restrict__ w,
                           const float* __restrict__ b, float* __restrict__ g1) {
  int i = blockIdx.x * blockDim.x + threadIdx.x;
  if (i >= NGRAPH * NHID) return;
  int g = i >> 6, j = i & 63;
  float s = b[j];
  for (int k = 0; k < HID2; ++k) s += pooled[g * HID2 + k] * w[k * NHID + j];
  g1[i] = selu_f(s);
}

__global__ void head_kernel(const float* __restrict__ g1, const float* __restrict__ w,
                            const float* __restrict__ b, float* __restrict__ out) {
  int g = blockIdx.x * blockDim.x + threadIdx.x;
  if (g >= NGRAPH) return;
  float z0 = b[0], z1 = b[1];
  for (int k = 0; k < NHID; ++k) {
    float v = g1[g * NHID + k];
    z0 += v * w[2 * k + 0];
    z1 += v * w[2 * k + 1];
  }
  float mx = fmaxf(z0, z1);
  float lse = mx + logf(expf(z0 - mx) + expf(z1 - mx));
  out[2 * g + 0] = z0 - lse;
  out[2 * g + 1] = z1 - lse;
}

extern "C" void kernel_launch(void* const* d_in, const int* in_sizes, int n_in,
                              void* d_out, int out_size, void* d_ws, size_t ws_size,
                              hipStream_t stream) {
  const float* x    = (const float*)d_in[0];
  const int*   ei   = (const int*)d_in[1];
  const int*   src  = ei;
  const int*   dstp = ei + NEDGES;
  const int*   batch= (const int*)d_in[2];
  const float* W1   = (const float*)d_in[3];
  const float* as1  = (const float*)d_in[4];
  const float* ad1  = (const float*)d_in[5];
  const float* b1   = (const float*)d_in[6];
  const float* W2   = (const float*)d_in[7];
  const float* as2  = (const float*)d_in[8];
  const float* ad2  = (const float*)d_in[9];
  const float* b2   = (const float*)d_in[10];
  const float* fc1w = (const float*)d_in[11];
  const float* fc1b = (const float*)d_in[12];
  const float* fc2w = (const float*)d_in[13];
  const float* fc2b = (const float*)d_in[14];
  float* out = (float*)d_out;

  char* w = (char*)d_ws;
  auto alloc = [&](size_t bytes) { void* p = w; w += (bytes + 255) & ~(size_t)255; return p; };
  float*    P    = (float*)alloc((size_t)NNODES * HID2 * 4);   // xw buffer
  float*    Q    = (float*)alloc((size_t)NNODES * HID2 * 4);   // aggregated h
  float*    es   = (float*)alloc((size_t)NNODES * 4);
  float*    ed   = (float*)alloc((size_t)NNODES * 4);
  unsigned* mmax = (unsigned*)alloc((size_t)NNODES * 4);
  float*    den  = (float*)alloc((size_t)NNODES * 4);
  float*    ee   = (float*)alloc((size_t)ETOT * 4);
  __bf16*   Wp1  = (__bf16*)alloc((size_t)INF_ * HID2 * 2);
  __bf16*   Wp2  = (__bf16*)alloc((size_t)HID2 * HID2 * 2);
  float*    sums = (float*)alloc((size_t)NGRAPH * HID2 * 4);
  float*    cnt  = (float*)alloc((size_t)NGRAPH * 4);
  float*    g1   = (float*)alloc((size_t)NGRAPH * NHID * 4);

  const int B = 256;
  const int nodeWaveGrid = (NNODES + 7) / 8;          // one wave per node
  const int edgeGrid     = (ETOT + B - 1) / B;
  const int edgeWaveGrid = (ETOT + 7) / 8;            // one wave per edge
  const int hGrid        = (NNODES * HID2 + B - 1) / B;
  const int gemmGrid     = ((NNODES / 16) + 7) / 8;   // 8 row-tiles per block
  const int nGrid        = (NNODES + B - 1) / B;

  // pack weights into WMMA-B bf16 layout
  pack_w_bf16<<<(INF_ / 32 * 256 + B - 1) / B, B, 0, stream>>>(W1, Wp1, INF_);
  pack_w_bf16<<<(HID2 / 32 * 256 + B - 1) / B, B, 0, stream>>>(W2, Wp2, HID2);

  for (int layer = 0; layer < 2; ++layer) {
    if (layer == 0)
      gemm_wmma_bf16<<<gemmGrid, B, 0, stream>>>(x, Wp1, P, NNODES, INF_);
    else
      gemm_wmma_bf16<<<gemmGrid, B, 0, stream>>>(Q, Wp2, P, NNODES, HID2);

    const float* as = layer ? as2 : as1;
    const float* ad = layer ? ad2 : ad1;
    const float* bb = layer ? b2 : b1;

    node_attn<<<nodeWaveGrid, B, 0, stream>>>(P, as, ad, es, ed, NNODES);
    fill_f32<<<hGrid, B, 0, stream>>>(Q, 0.f, NNODES * HID2);
    fill_f32<<<nGrid, B, 0, stream>>>((float*)mmax, 0.f, NNODES);
    fill_f32<<<nGrid, B, 0, stream>>>(den, 0.f, NNODES);
    edge_logits<<<edgeGrid, B, 0, stream>>>(es, ed, src, dstp, ee, mmax);
    edge_expsum<<<edgeGrid, B, 0, stream>>>(src, dstp, ee, mmax, den);
    edge_scatter<<<edgeWaveGrid, B, 0, stream>>>(src, dstp, ee, den, P, Q);
    node_bias_selu<<<hGrid, B, 0, stream>>>(Q, bb, NNODES * HID2);
  }

  fill_f32<<<(NGRAPH * HID2 + B - 1) / B, B, 0, stream>>>(sums, 0.f, NGRAPH * HID2);
  fill_f32<<<1, B, 0, stream>>>(cnt, 0.f, NGRAPH);
  pool_sum<<<nodeWaveGrid, B, 0, stream>>>(Q, batch, sums, cnt, NNODES);
  pool_final<<<(NGRAPH * HID2 + B - 1) / B, B, 0, stream>>>(sums, cnt);
  fc1_kernel<<<(NGRAPH * NHID + B - 1) / B, B, 0, stream>>>(sums, fc1w, fc1b, g1);
  head_kernel<<<1, NGRAPH, 0, stream>>>(g1, fc2w, fc2b, out);
}
